// BaseModel_28132035789326
// MI455X (gfx1250) — matrix operand order repaired and checked by reading, not compiled, not run on hardware
//
#include <hip/hip_runtime.h>
#include <hip/hip_bf16.h>

#define EMB     256
#define FFN_D   512
#define READ_F  24
#define READ_E  128
#define INFO_F  10
#define INFO_E  64
#define SEQ_E   64
#define REF_LEN 25
#define CONV_C  32
#define KERN    3
#define CONV_L  (REF_LEN - KERN + 1)      // 23
#define CONV_FLAT (CONV_C * CONV_L)       // 736
#define AGG_H   256
#define AGG_OUT 128

typedef __attribute__((ext_vector_type(16))) __bf16 v16bf;
typedef __attribute__((ext_vector_type(8)))  float  v8f;

union Frag16 {
  unsigned short u[16];
  uint4          q[2];
  v16bf          v;
};

__device__ __forceinline__ unsigned short f2bf(float f) {
  union { float f; unsigned u; } x; x.f = f;
  unsigned r = x.u + 0x7FFFu + ((x.u >> 16) & 1u);   // round-to-nearest-even
  return (unsigned short)(r >> 16);
}
__device__ __forceinline__ float bf2f(unsigned short b) {
  union { unsigned u; float f; } x; x.u = ((unsigned)b) << 16;
  return x.f;
}

// Convert 8 contiguous f32 (two b128 loads) into 8 bf16 fragment slots.
__device__ __forceinline__ void pack8(Frag16& a, int off, const float* __restrict__ src) {
  float4 f0 = *(const float4*)(src);
  float4 f1 = *(const float4*)(src + 4);
  a.u[off + 0] = f2bf(f0.x); a.u[off + 1] = f2bf(f0.y);
  a.u[off + 2] = f2bf(f0.z); a.u[off + 3] = f2bf(f0.w);
  a.u[off + 4] = f2bf(f1.x); a.u[off + 5] = f2bf(f1.y);
  a.u[off + 6] = f2bf(f1.z); a.u[off + 7] = f2bf(f1.w);
}

// A-fragment (16x32 bf16) from row-major f32 [M, lda]; per ISA 7.12.2:
// lane<16: row M=lane, K = {k0..k0+7} U {k0+16..k0+23}; lane>=16: +8 on both groups.
// 8-element K-groups are contiguous: fast path = two float4 loads per group.
__device__ __forceinline__ Frag16 loadA_f32(const float* __restrict__ A, int lda,
                                            int m0, int Mtot, int k0, int Ka, int lane) {
  Frag16 a;
  int row = m0 + (lane & 15);
  if (row >= Mtot) row = Mtot - 1;
  const float* Ar = A + (long)row * lda;
  int kh = (lane & 16) ? 8 : 0;
  int kA = k0 + kh, kB = k0 + 16 + kh;
  if (kA + 7 < Ka) {
    pack8(a, 0, Ar + kA);
  } else {
#pragma unroll
    for (int e = 0; e < 8; ++e) a.u[e] = (kA + e < Ka) ? f2bf(Ar[kA + e]) : (unsigned short)0;
  }
  if (kB + 7 < Ka) {
    pack8(a, 8, Ar + kB);
  } else {
#pragma unroll
    for (int e = 0; e < 8; ++e) a.u[8 + e] = (kB + e < Ka) ? f2bf(Ar[kB + e]) : (unsigned short)0;
  }
  return a;
}

// A-fragment from LDS bf16 tile [16][ldh] (rows local 0..15, Kp full, 16B-aligned).
__device__ __forceinline__ Frag16 loadA_lds(const unsigned short* H, int ldh, int k0, int lane) {
  Frag16 a;
  const unsigned short* Hr = H + (lane & 15) * ldh;
  int kh = (lane & 16) ? 8 : 0;
  a.q[0] = *(const uint4*)(Hr + k0 + kh);
  a.q[1] = *(const uint4*)(Hr + k0 + 16 + kh);
  return a;
}

// B-fragment (32x16 bf16) from row-major bf16 [Kp, ldb]: lane holds column N=n0+lane%16,
// K = {k0..k0+7} U {k0+16..k0+23}; lane>=16: +8 on both groups (strided loads).
__device__ __forceinline__ Frag16 loadB(const unsigned short* __restrict__ B, int ldb,
                                        int k0, int n0, int lane) {
  Frag16 b;
  const unsigned short* Bc = B + n0 + (lane & 15);
  int kh = (lane & 16) ? 8 : 0;
#pragma unroll
  for (int e = 0; e < 8; ++e) {
    b.u[e]     = Bc[(long)(k0 + kh + e) * ldb];
    b.u[8 + e] = Bc[(long)(k0 + 16 + kh + e) * ldb];
  }
  return b;
}

#define WMMA_BF16(a, b, c) \
  __builtin_amdgcn_wmma_f32_16x16x32_bf16(false, (a).v, false, (b).v, (short)0, (c), false, false)

// ---------------------------------------------------------------------------
// Weight conversion: f32 [Ka,N] -> bf16 [Kp,N], zero-padded K rows.
__global__ void cvt_pad(const float* __restrict__ src, unsigned short* __restrict__ dst,
                        int Ka, int Kp, int N) {
  long idx = (long)blockIdx.x * blockDim.x + threadIdx.x;
  if (idx >= (long)Kp * N) return;
  int k = (int)(idx / N);
  dst[idx] = (k < Ka) ? f2bf(src[idx]) : (unsigned short)0;
}

// Serial prefix scans of ref/alt counts (once per launch; V=4096, negligible).
__global__ void scan_counts(const int* __restrict__ rc, const int* __restrict__ ac,
                            int* __restrict__ roff, int* __restrict__ aoff, int Vn) {
  if (threadIdx.x == 0 && blockIdx.x == 0) {
    int s = 0;
    for (int v = 0; v < Vn; ++v) { roff[v] = s; s += rc[v]; }
    roff[Vn] = s;
    s = 0;
    for (int v = 0; v < Vn; ++v) { aoff[v] = s; s += ac[v]; }
    aoff[Vn] = s;
  }
}

__global__ void fill_rowvar(const int* __restrict__ roff, const int* __restrict__ aoff,
                            int* __restrict__ rowvar, int Rref, int Vn) {
  int v = blockIdx.x * blockDim.x + threadIdx.x;
  if (v >= Vn) return;
  for (int r = roff[v]; r < roff[v + 1]; ++r) rowvar[r] = v;
  for (int r = aoff[v]; r < aoff[v + 1]; ++r) rowvar[Rref + r] = v;
}

// ---------------------------------------------------------------------------
// Fused read MLP: X[:,0:128] = relu(reads[R,24] @ W1 + b1) @ W2 + b2
// 4 waves/block, one 16-row tile per wave, hidden staged in LDS as bf16.
__global__ void read_mlp_fused(const float* __restrict__ reads,
                               const unsigned short* __restrict__ W1b,   // [32,128] padded
                               const float* __restrict__ b1,
                               const unsigned short* __restrict__ W2b,   // [128,128]
                               const float* __restrict__ b2,
                               float* __restrict__ X, int R) {
  __shared__ unsigned short hbuf[4][16][READ_E];   // 16 KB
  int lane = threadIdx.x & 31;
  int wave = threadIdx.x >> 5;
  int m0 = (blockIdx.x * 4 + wave) << 4;
  bool active = (m0 < R);
  int mbase = m0 + ((lane & 16) ? 8 : 0);
  int ml = (lane & 16) ? 8 : 0;

  if (active) {
    Frag16 a = loadA_f32(reads, READ_F, m0, R, 0, READ_F, lane);   // Ka=24, padded to 32
#pragma unroll
    for (int nt = 0; nt < READ_E / 16; ++nt) {
      v8f acc = {0, 0, 0, 0, 0, 0, 0, 0};
      Frag16 b = loadB(W1b, READ_E, 0, nt * 16, lane);
      acc = WMMA_BF16(a, b, acc);
      int ncol = nt * 16 + (lane & 15);
      float bv = b1[ncol];
#pragma unroll
      for (int i = 0; i < 8; ++i) {
        float v = acc[i] + bv;
        v = v > 0.f ? v : 0.f;
        hbuf[wave][ml + i][ncol] = f2bf(v);
      }
    }
  }
  __syncthreads();
  if (active) {
    // 2 groups of 4 n-tiles: one LDS A-fragment read feeds 4 WMMAs.
#pragma unroll
    for (int ntg = 0; ntg < 2; ++ntg) {
      v8f acc[4];
#pragma unroll
      for (int j = 0; j < 4; ++j) acc[j] = (v8f){0, 0, 0, 0, 0, 0, 0, 0};
#pragma unroll
      for (int k0 = 0; k0 < READ_E; k0 += 32) {
        Frag16 a = loadA_lds(&hbuf[wave][0][0], READ_E, k0, lane);
#pragma unroll
        for (int j = 0; j < 4; ++j) {
          Frag16 b = loadB(W2b, READ_E, k0, (ntg * 4 + j) * 16, lane);
          acc[j] = WMMA_BF16(a, b, acc[j]);
        }
      }
#pragma unroll
      for (int j = 0; j < 4; ++j) {
        int ncol = (ntg * 4 + j) * 16 + (lane & 15);
        float bv = b2[ncol];
#pragma unroll
        for (int i = 0; i < 8; ++i) {
          int m = mbase + i;
          if (m < R) X[(long)m * EMB + ncol] = acc[j][i] + bv;
        }
      }
    }
  }
}

// ---------------------------------------------------------------------------
// Per-variant info MLP + conv + seq linear -> IS[V,128] (info | seq). Tiny work.
__global__ void info_seq(const float* __restrict__ info,
                         const float* __restrict__ iw1, const float* __restrict__ ib1,
                         const float* __restrict__ iw2, const float* __restrict__ ib2,
                         const float* __restrict__ seq,
                         const float* __restrict__ cw, const float* __restrict__ cb,
                         const float* __restrict__ sw, const float* __restrict__ sb,
                         float* __restrict__ IS) {
  __shared__ float sinfo[INFO_F];
  __shared__ float sh[INFO_E];
  __shared__ float sconv[CONV_FLAT];
  __shared__ float sseq[4 * REF_LEN];
  int v = blockIdx.x, t = threadIdx.x;   // 64 threads
  if (t < INFO_F) sinfo[t] = info[v * INFO_F + t];
  for (int i = t; i < 4 * REF_LEN; i += 64) sseq[i] = seq[v * 4 * REF_LEN + i];
  __syncthreads();
  // info layer 1
  float h = ib1[t];
  for (int f = 0; f < INFO_F; ++f) h += sinfo[f] * iw1[f * INFO_E + t];
  sh[t] = h > 0.f ? h : 0.f;
  // conv (flatten index = c*CONV_L + p, matches reshape order)
  for (int idx = t; idx < CONV_FLAT; idx += 64) {
    int c = idx / CONV_L, p = idx % CONV_L;
    float s = cb[c];
#pragma unroll
    for (int i = 0; i < 4; ++i)
#pragma unroll
      for (int k = 0; k < KERN; ++k)
        s += sseq[i * REF_LEN + p + k] * cw[(c * 4 + i) * KERN + k];
    sconv[idx] = s > 0.f ? s : 0.f;
  }
  __syncthreads();
  float o1 = ib2[t];
  for (int f = 0; f < INFO_E; ++f) o1 += sh[f] * iw2[f * INFO_E + t];
  float o2 = sb[t];
  for (int f = 0; f < CONV_FLAT; ++f) o2 += sconv[f] * sw[f * SEQ_E + t];
  IS[v * 128 + t] = o1;
  IS[v * 128 + SEQ_E + t] = o2;
}

// Broadcast per-variant info|seq into X[:,128:256] per read.
__global__ void concat_is(const float* __restrict__ IS, const int* __restrict__ rowvar,
                          float* __restrict__ X, int R) {
  long idx = (long)blockIdx.x * blockDim.x + threadIdx.x;
  if (idx >= (long)R * 128) return;
  int r = (int)(idx >> 7), j = (int)(idx & 127);
  X[(long)r * EMB + 128 + j] = IS[rowvar[r] * 128 + j];
}

// Segment means over contiguous row ranges. means[0..V) = alt_mean, [V..2V) = ref_mean.
__global__ void seg_means(const float* __restrict__ X,
                          const int* __restrict__ roff, const int* __restrict__ aoff,
                          float* __restrict__ means, int Rref, int Vn) {
  int v = blockIdx.x, j = threadIdx.x;   // 256 threads
  int s, e, base;
  if (v < Vn) { s = aoff[v]; e = aoff[v + 1]; base = Rref; }
  else        { int vv = v - Vn; s = roff[vv]; e = roff[vv + 1]; base = 0; }
  float sum = 0.f;
  for (int r = s; r < e; ++r) sum += X[(long)(base + r) * EMB + j];
  int cnt = e - s;
  means[(long)v * EMB + j] = (cnt > 0) ? sum / (float)cnt : 0.f;
}

// ---------------------------------------------------------------------------
// Generic GEMM: out = act(A[M,Ka] @ Bw[Kp,N] + bias), A f32->bf16 on load,
// Bw pre-converted bf16. Each wave owns one 16-row tile x 4 n-tiles (N%64==0),
// so every A fragment feeds 4 WMMAs. act: 0=none, 1=relu, 2=sigmoid.
__global__ void gemm_bf16w(const float* __restrict__ A, int lda,
                           const unsigned short* __restrict__ Bw, int N, int Kp, int Ka,
                           const float* __restrict__ bias,
                           float* __restrict__ outF, unsigned short* __restrict__ outB,
                           int ldc, int M, int act) {
  int lane = threadIdx.x & 31;
  int wave = threadIdx.x >> 5;
  int tI = blockIdx.x * (blockDim.x >> 5) + wave;
  int ngroups = N >> 6;                      // groups of 4 n-tiles
  int mtiles = (M + 15) >> 4;
  if (tI >= ngroups * mtiles) return;        // wave-uniform
  int m0 = (tI / ngroups) << 4;
  int n0 = (tI % ngroups) << 6;
  v8f acc[4];
#pragma unroll
  for (int j = 0; j < 4; ++j) acc[j] = (v8f){0, 0, 0, 0, 0, 0, 0, 0};
  for (int k0 = 0; k0 < Kp; k0 += 32) {
    if (k0 + 32 < Kp)   // hide L2 latency for next K-step of weights
      __builtin_prefetch(Bw + (long)(k0 + 32 + lane) * N + n0, 0, 3);
    Frag16 a = loadA_f32(A, lda, m0, M, k0, Ka, lane);
#pragma unroll
    for (int j = 0; j < 4; ++j) {
      Frag16 b = loadB(Bw, N, k0, n0 + j * 16, lane);
      acc[j] = WMMA_BF16(a, b, acc[j]);
    }
  }
  int mbase = m0 + ((lane & 16) ? 8 : 0);
#pragma unroll
  for (int j = 0; j < 4; ++j) {
    int ncol = n0 + j * 16 + (lane & 15);
    float bv = bias ? bias[ncol] : 0.f;
#pragma unroll
    for (int i = 0; i < 8; ++i) {
      int m = mbase + i;
      float v = acc[j][i] + bv;
      if (act == 1)      v = v > 0.f ? v : 0.f;
      else if (act == 2) v = 1.f / (1.f + expf(-v));
      if (m < M) {
        if (outF) outF[(long)m * ldc + ncol] = v;
        else      outB[(long)m * ldc + ncol] = f2bf(v);
      }
    }
  }
}

// ---------------------------------------------------------------------------
// Fused gated-residual block (in place on X):
//   h = relu(X@W1+b1) * G[gaterow(m)]; X += h@W2 + b2
// 4 waves/block, 16-row tile per wave, h staged in LDS bf16 (64 KB).
// Phase 1: A tile pre-converted once into 8 register fragments, reused by all
// 32 n-tiles. Phase 2: 4 groups of 4 n-tiles; one LDS A read feeds 4 WMMAs.
__global__ void block_ffn(float* __restrict__ X,
                          const int* __restrict__ rowvar,
                          const unsigned short* __restrict__ W1b,  // [256,512] bf16
                          const float* __restrict__ b1,            // [512]
                          const unsigned short* __restrict__ G,    // [2V,512] bf16 gates
                          const unsigned short* __restrict__ W2b,  // [512,256] bf16
                          const float* __restrict__ b2,            // [256]
                          int R, int Rref, int Vn) {
  __shared__ unsigned short hbuf[4][16][FFN_D];   // 64 KB
  int lane = threadIdx.x & 31;
  int wave = threadIdx.x >> 5;
  int m0 = (blockIdx.x * 4 + wave) << 4;
  bool active = (m0 < R);
  int mbase = m0 + ((lane & 16) ? 8 : 0);
  int ml = (lane & 16) ? 8 : 0;

  if (active) {
    // preload + convert the 16x256 A tile once (8 K-step fragments, b128 loads)
    Frag16 afr[8];
#pragma unroll
    for (int kt = 0; kt < 8; ++kt)
      afr[kt] = loadA_f32(X, EMB, m0, R, kt * 32, EMB, lane);
    // gate row per output row of this lane
    int grow[8];
#pragma unroll
    for (int i = 0; i < 8; ++i) {
      int m = mbase + i; if (m >= R) m = R - 1;
      grow[i] = rowvar[m] + ((m >= Rref) ? Vn : 0);
    }
    for (int nt = 0; nt < FFN_D / 16; ++nt) {
      v8f acc = {0, 0, 0, 0, 0, 0, 0, 0};
#pragma unroll
      for (int kt = 0; kt < 8; ++kt) {
        if (nt + 1 < FFN_D / 16)
          __builtin_prefetch(W1b + (long)(kt * 32 + lane) * FFN_D + (nt + 1) * 16, 0, 3);
        Frag16 b = loadB(W1b, FFN_D, kt * 32, nt * 16, lane);
        acc = WMMA_BF16(afr[kt], b, acc);
      }
      int ncol = nt * 16 + (lane & 15);
      float bv = b1[ncol];
#pragma unroll
      for (int i = 0; i < 8; ++i) {
        float v = acc[i] + bv;
        v = v > 0.f ? v : 0.f;
        v *= bf2f(G[(long)grow[i] * FFN_D + ncol]);
        hbuf[wave][ml + i][ncol] = f2bf(v);
      }
    }
  }
  __syncthreads();
  if (active) {
#pragma unroll
    for (int ntg = 0; ntg < 4; ++ntg) {
      v8f acc[4];
#pragma unroll
      for (int j = 0; j < 4; ++j) acc[j] = (v8f){0, 0, 0, 0, 0, 0, 0, 0};
      for (int k0 = 0; k0 < FFN_D; k0 += 32) {
        Frag16 a = loadA_lds(&hbuf[wave][0][0], FFN_D, k0, lane);
#pragma unroll
        for (int j = 0; j < 4; ++j) {
          Frag16 b = loadB(W2b, EMB, k0, (ntg * 4 + j) * 16, lane);
          acc[j] = WMMA_BF16(a, b, acc[j]);
        }
      }
#pragma unroll
      for (int j = 0; j < 4; ++j) {
        int ncol = (ntg * 4 + j) * 16 + (lane & 15);
        float bv = b2[ncol];
#pragma unroll
        for (int i = 0; i < 8; ++i) {
          int m = mbase + i;
          if (m < R) {
            long idx = (long)m * EMB + ncol;
            X[idx] += acc[j][i] + bv;    // residual, row owned by this wave only
          }
        }
      }
    }
  }
}

// Normalized weighted segment sums over alt reads.
__global__ void alt_wmean(const float* __restrict__ X, const int* __restrict__ aoff,
                          const float* __restrict__ alt_u, float* __restrict__ AM,
                          int Rref, int Vn) {
  int v = blockIdx.x, j = threadIdx.x;   // 256 threads
  int s = aoff[v], e = aoff[v + 1];
  float wsum = 0.f;
  for (int r = s; r < e; ++r) wsum += 1.0f + 0.2f * (1.0f - 2.0f * alt_u[r]);
  float sum = 0.f;
  for (int r = s; r < e; ++r) {
    float w = 1.0f + 0.2f * (1.0f - 2.0f * alt_u[r]);
    sum += w * X[(long)(Rref + r) * EMB + j];
  }
  AM[(long)v * EMB + j] = (wsum != 0.f) ? sum / wsum : 0.f;
}

// ---------------------------------------------------------------------------
extern "C" void kernel_launch(void* const* d_in, const int* in_sizes, int n_in,
                              void* d_out, int out_size, void* d_ws, size_t ws_size,
                              hipStream_t stream) {
  (void)n_in; (void)out_size; (void)ws_size;
  const float* reads = (const float*)d_in[0];
  const float* info  = (const float*)d_in[1];
  const float* seqoh = (const float*)d_in[2];
  const float* altu  = (const float*)d_in[3];
  const int*   rc    = (const int*)d_in[4];
  const int*   ac    = (const int*)d_in[5];
  const float* rw1 = (const float*)d_in[6],  *rb1 = (const float*)d_in[7];
  const float* rw2 = (const float*)d_in[8],  *rb2 = (const float*)d_in[9];
  const float* iw1 = (const float*)d_in[10], *ib1 = (const float*)d_in[11];
  const float* iw2 = (const float*)d_in[12], *ib2 = (const float*)d_in[13];
  const float* cw  = (const float*)d_in[14], *cb  = (const float*)d_in[15];
  const float* sw  = (const float*)d_in[16], *sb  = (const float*)d_in[17];
  const float* bw1 = (const float*)d_in[18], *bb1 = (const float*)d_in[19];
  const float* bwg = (const float*)d_in[20], *bbg = (const float*)d_in[21];
  const float* bw2 = (const float*)d_in[22], *bb2 = (const float*)d_in[23];
  const float* aw1 = (const float*)d_in[24], *ab1 = (const float*)d_in[25];
  const float* aw2 = (const float*)d_in[26], *ab2 = (const float*)d_in[27];

  int R    = in_sizes[0] / READ_F;
  int A    = in_sizes[3];
  int Vn   = in_sizes[1] / INFO_F;
  int Rref = R - A;

  // ---- workspace partition (hip allocations are 256B aligned) ----
  char* p = (char*)d_ws;
  auto alloc = [&](size_t bytes) -> void* {
    void* q = (void*)p;
    p += (bytes + 255) & ~(size_t)255;
    return q;
  };
  int*   roff  = (int*)alloc((Vn + 1) * sizeof(int));
  int*   aoff  = (int*)alloc((Vn + 1) * sizeof(int));
  int*   rowvar= (int*)alloc((size_t)R * sizeof(int));
  float* IS    = (float*)alloc((size_t)Vn * 128 * sizeof(float));
  float* X     = (float*)alloc((size_t)R * EMB * sizeof(float));
  float* means = (float*)alloc((size_t)2 * Vn * EMB * sizeof(float));
  unsigned short* G = (unsigned short*)alloc((size_t)2 * Vn * FFN_D * sizeof(short));
  float* AM    = (float*)alloc((size_t)Vn * EMB * sizeof(float));
  float* AH    = (float*)alloc((size_t)Vn * AGG_H * sizeof(float));
  unsigned short* rw1b = (unsigned short*)alloc(32 * READ_E * 2);
  unsigned short* rw2b = (unsigned short*)alloc(READ_E * READ_E * 2);
  unsigned short* w1b  = (unsigned short*)alloc(2 * EMB * FFN_D * 2);
  unsigned short* wgb  = (unsigned short*)alloc(2 * EMB * FFN_D * 2);
  unsigned short* w2b  = (unsigned short*)alloc(2 * FFN_D * EMB * 2);
  unsigned short* aw1b = (unsigned short*)alloc(EMB * AGG_H * 2);
  unsigned short* aw2b = (unsigned short*)alloc(AGG_H * AGG_OUT * 2);

  auto cvt = [&](const float* s, unsigned short* d, int Ka, int Kp, int N) {
    long tot = (long)Kp * N;
    cvt_pad<<<(unsigned)((tot + 255) / 256), 256, 0, stream>>>(s, d, Ka, Kp, N);
  };
  cvt(rw1, rw1b, READ_F, 32, READ_E);
  cvt(rw2, rw2b, READ_E, READ_E, READ_E);
  cvt(bw1,               w1b,               EMB,   EMB,   FFN_D);
  cvt(bw1 + EMB * FFN_D, w1b + EMB * FFN_D, EMB,   EMB,   FFN_D);
  cvt(bwg,               wgb,               EMB,   EMB,   FFN_D);
  cvt(bwg + EMB * FFN_D, wgb + EMB * FFN_D, EMB,   EMB,   FFN_D);
  cvt(bw2,               w2b,               FFN_D, FFN_D, EMB);
  cvt(bw2 + FFN_D * EMB, w2b + FFN_D * EMB, FFN_D, FFN_D, EMB);
  cvt(aw1, aw1b, EMB, EMB, AGG_H);
  cvt(aw2, aw2b, AGG_H, AGG_H, AGG_OUT);

  scan_counts<<<1, 32, 0, stream>>>(rc, ac, roff, aoff, Vn);
  fill_rowvar<<<(Vn + 255) / 256, 256, 0, stream>>>(roff, aoff, rowvar, Rref, Vn);

  // Per-read embedding -> X[:,0:128]
  read_mlp_fused<<<(R + 63) / 64, 128, 0, stream>>>(reads, rw1b, rb1, rw2b, rb2, X, R);
  // Per-variant info|seq embedding, broadcast to X[:,128:256]
  info_seq<<<Vn, 64, 0, stream>>>(info, iw1, ib1, iw2, ib2, seqoh, cw, cb, sw, sb, IS);
  {
    long tot = (long)R * 128;
    concat_is<<<(unsigned)((tot + 255) / 256), 256, 0, stream>>>(IS, rowvar, X, R);
  }

  auto gemm = [&](const float* Ap, int lda, const unsigned short* Bp, int N, int Kp, int Ka,
                  const float* bias, float* oF, unsigned short* oB, int ldc, int M, int act) {
    int tiles = ((M + 15) / 16) * (N / 64);
    gemm_bf16w<<<(tiles + 3) / 4, 128, 0, stream>>>(Ap, lda, Bp, N, Kp, Ka, bias,
                                                    oF, oB, ldc, M, act);
  };

  for (int i = 0; i < 2; ++i) {
    // segment means (contiguous ranges): means[0..V)=alt_mean, [V..2V)=ref_mean
    seg_means<<<2 * Vn, EMB, 0, stream>>>(X, roff, aoff, means, Rref, Vn);
    // per-variant gates: G = sigmoid(means @ wg + bg), bf16, [2V,512]
    gemm(means, EMB, wgb + (size_t)i * EMB * FFN_D, FFN_D, EMB, EMB,
         bbg + i * FFN_D, nullptr, G, FFN_D, 2 * Vn, 2);
    // fused gated FFN, residual in place
    block_ffn<<<(R + 63) / 64, 128, 0, stream>>>(
        X, rowvar, w1b + (size_t)i * EMB * FFN_D, bb1 + i * FFN_D, G,
        w2b + (size_t)i * FFN_D * EMB, bb2 + i * EMB, R, Rref, Vn);
  }

  // normalized weighted alt-segment means
  alt_wmean<<<Vn, EMB, 0, stream>>>(X, aoff, altu, AM, Rref, Vn);
  // aggregation MLP -> d_out [V,128]
  gemm(AM, EMB, aw1b, AGG_H, EMB, EMB, ab1, AH, nullptr, AGG_H, Vn, 1);
  gemm(AH, AGG_H, aw2b, AGG_OUT, AGG_H, AGG_H, ab2, (float*)d_out, nullptr, AGG_OUT, Vn, 0);
}
